// Head_5385888989914
// MI455X (gfx1250) — matrix operand order repaired and checked by reading, not compiled, not run on hardware
//
#include <hip/hip_runtime.h>
#include <hip/hip_bf16.h>

// Problem dims (fixed by reference)
#define BB 8
#define TT 2048
#define CC 1024
#define HH 64
#define BT (BB * TT)   // 16384 rows

typedef __attribute__((ext_vector_type(16))) __bf16     v16bf;
typedef __attribute__((ext_vector_type(8)))  float      v8f;
typedef __attribute__((ext_vector_type(8)))  unsigned   v8u;

static __device__ __forceinline__ unsigned short f2bf(float f) {
    unsigned u = __float_as_uint(f);
    u += 0x7FFFu + ((u >> 16) & 1u);     // round-to-nearest-even
    return (unsigned short)(u >> 16);
}

static __device__ __forceinline__ v8f wmma_bf16(v8u a, v8u b, v8f c) {
    return __builtin_amdgcn_wmma_f32_16x16x32_bf16(
        /*neg_a=*/false, __builtin_bit_cast(v16bf, a),
        /*neg_b=*/false, __builtin_bit_cast(v16bf, b),
        /*c_mod=*/(short)0, c, /*reuse_a=*/false, /*reuse_b=*/false);
}

// ---------------- Kernel 1: x f32 -> bf16 (flat, float4 vectorized) ----------------
__global__ __launch_bounds__(256) void cvt_x(const float* __restrict__ x,
                                             unsigned short* __restrict__ xb, int n4) {
    int i = blockIdx.x * blockDim.x + threadIdx.x;
    if (i >= n4) return;
    const float4 v = ((const float4*)x)[i];
    unsigned short* o = xb + (size_t)i * 4;
    o[0] = f2bf(v.x); o[1] = f2bf(v.y); o[2] = f2bf(v.z); o[3] = f2bf(v.w);
}

// ---------------- Kernel 2: W [C,H] f32 -> Wt bf16 [3][H][C] (transposed) ----------
// slot 0 = Wq, 1 = Wk, 2 = Wv
__global__ __launch_bounds__(256) void cvt_w(const float* __restrict__ Wq,
                                             const float* __restrict__ Wk,
                                             const float* __restrict__ Wv,
                                             unsigned short* __restrict__ Wt) {
    int i = blockIdx.x * blockDim.x + threadIdx.x;           // over 3*H*C
    if (i >= 3 * HH * CC) return;
    int which = i / (HH * CC);
    int rem   = i - which * (HH * CC);
    int h = rem / CC;
    int c = rem - h * CC;
    const float* W = (which == 0) ? Wq : (which == 1) ? Wk : Wv;
    Wt[i] = f2bf(W[(size_t)c * HH + h]);
}

// ---------------- Kernel 3: fused QKV projection via WMMA --------------------------
// One wave computes a 16x64 tile of ALL THREE projections, sharing one A-fragment
// per 12 WMMAs (x is read once instead of three times).
__global__ __launch_bounds__(256) void qkv_proj(const unsigned short* __restrict__ xb,   // [BT][C]
                                                const unsigned short* __restrict__ Wt,   // [3][H][C]
                                                unsigned short* __restrict__ qb,         // [BT][H]
                                                unsigned short* __restrict__ kb,         // [BT][H]
                                                unsigned short* __restrict__ vTb) {      // [B][H][T]
    const int lane = threadIdx.x & 31;
    const int lh   = lane & 15;
    const int hi   = lane >> 4;                 // 0: lanes 0-15, 1: lanes 16-31
    const int wid  = blockIdx.x * 8 + (threadIdx.x >> 5);
    const int m_base = wid * 16;

    v8f acc[3][4];
#pragma unroll
    for (int q = 0; q < 3; ++q)
#pragma unroll
        for (int nt = 0; nt < 4; ++nt) acc[q][nt] = v8f{};

    const unsigned short* xrow = xb + (size_t)(m_base + lh) * CC;
    for (int c0 = 0; c0 < CC; c0 += 32) {
        // A fragment: 16x32 bf16, lane = row M, K pairs per ISA layout
        v8u a;
#pragma unroll
        for (int j = 0; j < 8; ++j) {
            int k = 2 * (j & 3) + 8 * hi + 16 * (j >> 2);
            a[j] = *(const unsigned*)(xrow + c0 + k);
        }
#pragma unroll
        for (int q = 0; q < 3; ++q) {
            const unsigned short* W = Wt + (size_t)q * HH * CC;
#pragma unroll
            for (int nt = 0; nt < 4; ++nt) {
                // B fragment: 32x16, element (K=c, N=h) = Wt[h][c]; VGPR j: K=2*(j&3)+8*hi+16*(j>>2)
                const unsigned short* wrow = W + (size_t)(nt * 16 + lh) * CC + c0 + 8 * hi;
                v8u bf;
#pragma unroll
                for (int j = 0; j < 8; ++j) bf[j] = *(const unsigned*)(wrow + 2 * (j & 3) + 16 * (j >> 2));
                acc[q][nt] = wmma_bf16(a, bf, acc[q][nt]);
            }
        }
    }

    // q: fold C^-0.5 = 1/32 (exact power of two) before bf16 quantization
#pragma unroll
    for (int nt = 0; nt < 4; ++nt)
#pragma unroll
        for (int r = 0; r < 8; ++r) {
            int m = m_base + r + 8 * hi;
            int h = nt * 16 + lh;
            qb[(size_t)m * HH + h] = f2bf(acc[0][nt][r] * 0.03125f);
            kb[(size_t)m * HH + h] = f2bf(acc[1][nt][r]);
            int b = m >> 11;             // /T
            int t = m & (TT - 1);
            vTb[((size_t)b * HH + h) * TT + t] = f2bf(acc[2][nt][r]);   // v transposed
        }
}

// ---------------- Kernel 4: flash attention (one wave = 16 query rows) -------------
__global__ __launch_bounds__(256) void attn(const unsigned short* __restrict__ qb,   // [BT][H], pre-scaled
                                            const unsigned short* __restrict__ kb,   // [BT][H]
                                            const unsigned short* __restrict__ vTb,  // [B][H][T]
                                            float* __restrict__ out) {               // [B][T][H]
    __shared__ __align__(16) unsigned short plds[8][16][32];   // per-wave P tile (bf16)

    const int lane = threadIdx.x & 31;
    const int lh   = lane & 15;
    const int hi   = lane >> 4;
    const int w    = threadIdx.x >> 5;
    const int wid  = blockIdx.x * 8 + w;
    const int b      = wid >> 7;             // T/16 = 128 strips per batch
    const int t_base = (wid & 127) << 4;

    // Load q A-fragments for h in [0,32) and [32,64)
    v8u qf[2];
    const unsigned short* qrow = qb + ((size_t)b * TT + t_base + lh) * HH;
#pragma unroll
    for (int p = 0; p < 2; ++p)
#pragma unroll
        for (int j = 0; j < 8; ++j) {
            int k = 2 * (j & 3) + 8 * hi + 16 * (j >> 2) + 32 * p;
            qf[p][j] = *(const unsigned*)(qrow + k);
        }

    v8f acc[4] = {v8f{}, v8f{}, v8f{}, v8f{}};
    float mrow[8], lrow[8];
#pragma unroll
    for (int r = 0; r < 8; ++r) { mrow[r] = -1e30f; lrow[r] = 0.0f; }

    const unsigned short* kb0 = kb + (size_t)b * TT * HH;
    const unsigned short* vb0 = vTb + (size_t)b * HH * TT;

    const int s_end = t_base + 16;           // exclusive bound on keys needed
    for (int s0 = 0; s0 < s_end; s0 += 32) {
        // ---- prefetch next iteration's k/v tiles (global_prefetch_b8) ----
        if (s0 + 32 < s_end) {
            // each k row is H*2 = 128B = exactly one cacheline; 32 lanes cover the tile
            __builtin_prefetch(kb0 + (size_t)(s0 + 32 + lane) * HH, 0, 3);
            // vT tile: 64 rows x 64B; two prefetches cover all rows' lines
            __builtin_prefetch(vb0 + (size_t)lane * TT + s0 + 32, 0, 3);
            __builtin_prefetch(vb0 + (size_t)(lane + 32) * TT + s0 + 32, 0, 3);
        }
        // ---- scores: S[16x32] = q(16x64) . k^T ----
        v8f sc[2] = {v8f{}, v8f{}};
        const unsigned short* kbase = kb0 + (size_t)s0 * HH;
#pragma unroll
        for (int nt = 0; nt < 2; ++nt) {
#pragma unroll
            for (int p = 0; p < 2; ++p) {
                const unsigned short* krow = kbase + (size_t)(nt * 16 + lh) * HH + 32 * p + 8 * hi;
                v8u bf;
#pragma unroll
                for (int j = 0; j < 8; ++j) bf[j] = *(const unsigned*)(krow + 2 * (j & 3) + 16 * (j >> 2));
                sc[nt] = wmma_bf16(qf[p], bf, sc[nt]);
            }
        }
        // ---- causal mask (only on partial diagonal tiles; uniform branch per wave) ----
        if (s0 + 31 > t_base) {
#pragma unroll
            for (int nt = 0; nt < 2; ++nt) {
                int s = s0 + nt * 16 + lh;
#pragma unroll
                for (int r = 0; r < 8; ++r) {
                    int t = t_base + r + 8 * hi;
                    sc[nt][r] = (s > t) ? -1e30f : sc[nt][r];
                }
            }
        }
        // ---- online softmax: row max across the 16-lane N-group ----
        float alpha[8];
#pragma unroll
        for (int r = 0; r < 8; ++r) {
            float v = fmaxf(sc[0][r], sc[1][r]);
            v = fmaxf(v, __shfl_xor(v, 1));
            v = fmaxf(v, __shfl_xor(v, 2));
            v = fmaxf(v, __shfl_xor(v, 4));
            v = fmaxf(v, __shfl_xor(v, 8));
            float mn = fmaxf(mrow[r], v);
            alpha[r] = __expf(mrow[r] - mn);
            mrow[r]  = mn;
        }
#pragma unroll
        for (int nt = 0; nt < 4; ++nt)
#pragma unroll
            for (int r = 0; r < 8; ++r) acc[nt][r] *= alpha[r];
        // ---- P = exp(S - m); row sums; stage P (bf16) to this wave's LDS tile ----
#pragma unroll
        for (int nt = 0; nt < 2; ++nt)
#pragma unroll
            for (int r = 0; r < 8; ++r) {
                float p = __expf(sc[nt][r] - mrow[r]);
                sc[nt][r] = p;
                plds[w][r + 8 * hi][nt * 16 + lh] = f2bf(p);
            }
#pragma unroll
        for (int r = 0; r < 8; ++r) {
            float v = sc[0][r] + sc[1][r];
            v += __shfl_xor(v, 1);
            v += __shfl_xor(v, 2);
            v += __shfl_xor(v, 4);
            v += __shfl_xor(v, 8);
            lrow[r] = lrow[r] * alpha[r] + v;
        }
        // ---- reload P as A-fragment (C-layout -> A-layout transpose via LDS) ----
        // Same-wave store->load: compiler inserts s_wait_dscnt; no block barrier needed.
        v8u pf;
#pragma unroll
        for (int j = 0; j < 8; ++j) {
            int k = 2 * (j & 3) + 8 * hi + 16 * (j >> 2);
            pf[j] = *(const unsigned*)&plds[w][lh][k];
        }
        // ---- out += P(16x32) . V(32x64);  B element (K=s, N=h) = vT[b][h][s0+K] ----
        const unsigned short* vbase = vb0 + s0 + 8 * hi;
#pragma unroll
        for (int nt = 0; nt < 4; ++nt) {
            const unsigned short* vrow = vbase + (size_t)(nt * 16 + lh) * TT;
            v8u vf;
#pragma unroll
            for (int j = 0; j < 8; ++j) vf[j] = *(const unsigned*)(vrow + 2 * (j & 3) + 16 * (j >> 2));
            acc[nt] = wmma_bf16(pf, vf, acc[nt]);
        }
    }

    // ---- normalize and store f32 output ----
#pragma unroll
    for (int nt = 0; nt < 4; ++nt)
#pragma unroll
        for (int r = 0; r < 8; ++r) {
            int t = t_base + r + 8 * hi;
            int h = nt * 16 + lh;
            out[((size_t)b * TT + t) * HH + h] = acc[nt][r] / lrow[r];
        }
}

// ---------------- host launcher ----------------------------------------------------
extern "C" void kernel_launch(void* const* d_in, const int* in_sizes, int n_in,
                              void* d_out, int out_size, void* d_ws, size_t ws_size,
                              hipStream_t stream) {
    (void)in_sizes; (void)n_in; (void)out_size; (void)ws_size;
    const float* x  = (const float*)d_in[0];
    const float* Wk = (const float*)d_in[1];
    const float* Wq = (const float*)d_in[2];
    const float* Wv = (const float*)d_in[3];
    float* out = (float*)d_out;

    char* ws = (char*)d_ws;
    size_t off = 0;
    unsigned short* xb  = (unsigned short*)(ws + off); off += (size_t)BT * CC * 2;      // 33.5 MB
    unsigned short* Wt  = (unsigned short*)(ws + off); off += (size_t)3 * HH * CC * 2;  // 384 KB
    unsigned short* qb  = (unsigned short*)(ws + off); off += (size_t)BT * HH * 2;      // 2 MB
    unsigned short* kb  = (unsigned short*)(ws + off); off += (size_t)BT * HH * 2;      // 2 MB
    unsigned short* vTb = (unsigned short*)(ws + off); off += (size_t)BT * HH * 2;      // 2 MB

    const int n4 = (BT * CC) / 4;
    cvt_x<<<(n4 + 255) / 256, 256, 0, stream>>>(x, xb, n4);
    cvt_w<<<(3 * HH * CC + 255) / 256, 256, 0, stream>>>(Wq, Wk, Wv, Wt);

    qkv_proj<<<BT / 16 / 8, 256, 0, stream>>>(xb, Wt, qb, kb, vTb);

    attn<<<BT / 16 / 8, 256, 0, stream>>>(qb, kb, vTb, out);
}
